// BioXNetLayer_4827543241281
// MI455X (gfx1250) — matrix operand order repaired and checked by reading, not compiled, not run on hardware
//
#include <hip/hip_runtime.h>
#include <hip/hip_bf16.h>

// ---- problem constants (match reference) ----
#define BN      128        // batch
#define GG      20000      // genes (K)
#define PP      4096       // pathways (N)
#define KT      32         // K tile = WMMA bf16 K
#define NT      32         // N tile per block
#define NIT     (GG / KT)  // 625 K-steps
#define BN_EPS  1e-5f

// output layout: [outcome B*P][decision B][attention B*P]
#define OUT_DEC_OFF  (BN * PP)
#define OUT_ATT_OFF  (BN * PP + BN)

// workspace layout: [wsdec (P/NT)*B floats][inputs as bf16, B*G ushorts]
#define WSDEC_ELEMS  ((PP / NT) * BN)

typedef __attribute__((ext_vector_type(16))) __bf16   v16bf;
typedef __attribute__((ext_vector_type(2)))  __bf16   v2bf;
typedef __attribute__((ext_vector_type(8)))  float    v8f;
typedef __attribute__((ext_vector_type(8)))  unsigned v8u;
typedef __attribute__((ext_vector_type(4)))  unsigned u4v;

union BfCast { v8u u; v16bf b; };

#if defined(__has_builtin)
#  if __has_builtin(__builtin_amdgcn_cvt_pk_bf16_f32)
#    define HAVE_PK_BF16 1
#  endif
#endif

__device__ __forceinline__ unsigned pack2bf(float lo, float hi) {
#ifdef HAVE_PK_BF16
    v2bf p = __builtin_amdgcn_cvt_pk_bf16_f32(lo, hi);     // v_cvt_pk_bf16_f32
    return __builtin_bit_cast(unsigned, p);
#else
    // let clang pick the native fptrunc lowering if gfx1250 has one
    v2bf p;
    p[0] = (__bf16)lo;
    p[1] = (__bf16)hi;
    return __builtin_bit_cast(unsigned, p);
#endif
}

__device__ __forceinline__ float fast_sigmoid(float x) {
    return 1.0f / (1.0f + __expf(-x));
}
__device__ __forceinline__ float fast_tanh(float x) {
    x = fminf(15.0f, fmaxf(-15.0f, x));
    float e = __expf(2.0f * x);
    return (e - 1.0f) / (e + 1.0f);
}

// ---- one-time conversion: inputs f32 -> bf16 (shared by all GEMM blocks) ----
__global__ void __launch_bounds__(256)
prep_inputs_bf16(const float* __restrict__ inp, unsigned* __restrict__ dst)
{
    size_t i = (size_t)blockIdx.x * blockDim.x + threadIdx.x;  // float4 index
    if (i >= (size_t)BN * GG / 4) return;
    float4 v = ((const float4*)inp)[i];
    dst[2 * i]     = pack2bf(v.x, v.y);
    dst[2 * i + 1] = pack2bf(v.z, v.w);
}

__global__ void __launch_bounds__(256)
bioxnet_fused(const unsigned short* __restrict__ abf,  // [B,G] bf16 (prepped)
              const float* __restrict__ mapp,  // [G,P]
              const float* __restrict__ ker,   // [G,P]
              const float* __restrict__ bias,  // [P]
              const float* __restrict__ attk,  // [G,P]
              const float* __restrict__ attb,  // [P]
              const float* __restrict__ gamma, // [P]
              const float* __restrict__ beta,  // [P]
              const float* __restrict__ decw,  // [P]
              float*       __restrict__ out,   // [B*P + B + B*P]
              float*       __restrict__ wsdec) // [P/NT, B] partial decisions
{
    // ---- LDS (~45 KB total), double-buffered weight tiles ----
    __shared__ __align__(16) unsigned short bS[2][NT * KT]; // masked kernel bf16 [n][k]
    __shared__ __align__(16) unsigned short cS[2][NT * KT]; // att kernel bf16 [n][k]
    __shared__ float hS[BN * NT];
    __shared__ float tS[BN * NT];
    __shared__ float meanS[NT];
    __shared__ float rstdS[NT];
    __shared__ float rowpart[256];

    const int tid  = threadIdx.x;
    const int wave = tid >> 5;            // 0..7 -> batch rows [16w, 16w+16)
    const int lane = tid & 31;
    const int p0   = blockIdx.x * NT;

    // weights staging: one float4 per thread per matrix
    const int wk = tid >> 3;
    const int wn = (tid & 7) * 4;
    const float4* wp = (const float4*)(ker  + (size_t)wk * PP + p0 + wn);
    const float4* mp = (const float4*)(mapp + (size_t)wk * PP + p0 + wn);
    const float4* tp = (const float4*)(attk + (size_t)wk * PP + p0 + wn);
    const int WSTEP = KT * PP / 4;        // float4 step per K-tile

    // A fragment: direct global loads from bf16 inputs, contiguous k order (k = 16*(lane/16) + s).
    const int m_frag = wave * 16 + (lane & 15);
    const int kh     = (lane >> 4) * 16;
    const u4v* apl = (const u4v*)(abf + (size_t)m_frag * GG + kh);  // 16B aligned

    // B fragment LDS coords (invariant)
    const int n_low = lane & 15;
    const int kb    = (lane >> 4) * 8;    // dwords: lanes 16-31 hold K=16..31

    v8f accH[2]; v8f accA[2];
    accH[0] = (v8f)0.f; accH[1] = (v8f)0.f;
    accA[0] = (v8f)0.f; accA[1] = (v8f)0.f;

    // ---- prologue: stage weight tile 0 into buffer 0 ----
    {
        float4 w = wp[0], m = mp[0], t = tp[0];
        unsigned k0p = pack2bf(w.x * m.x, w.y * m.y);
        unsigned k1p = pack2bf(w.z * m.z, w.w * m.w);
        unsigned t0p = pack2bf(t.x, t.y);
        unsigned t1p = pack2bf(t.z, t.w);
        bS[0][(wn + 0) * KT + wk] = (unsigned short)k0p;
        bS[0][(wn + 1) * KT + wk] = (unsigned short)(k0p >> 16);
        bS[0][(wn + 2) * KT + wk] = (unsigned short)k1p;
        bS[0][(wn + 3) * KT + wk] = (unsigned short)(k1p >> 16);
        cS[0][(wn + 0) * KT + wk] = (unsigned short)t0p;
        cS[0][(wn + 1) * KT + wk] = (unsigned short)(t0p >> 16);
        cS[0][(wn + 2) * KT + wk] = (unsigned short)t1p;
        cS[0][(wn + 3) * KT + wk] = (unsigned short)(t1p >> 16);
    }
    __syncthreads();

    // ---- pipelined main loop: ONE barrier per iteration (double buffer) ----
    for (int it = 0; it < NIT; ++it) {
        const int  cur  = it & 1;
        const int  nxt  = cur ^ 1;
        const bool more = (it + 1 < NIT);

        float4 nw, nm, nt4;
        if (more) {
            size_t o = (size_t)(it + 1) * WSTEP;
            nw = wp[o]; nm = mp[o]; nt4 = tp[o];
            if (it + 2 < NIT) {                         // distance-2 L2 prefetch
                __builtin_prefetch(&wp[(size_t)(it + 2) * WSTEP], 0, 1);
                __builtin_prefetch(&tp[(size_t)(it + 2) * WSTEP], 0, 1);
            }
        }

        // A fragment straight from global (bf16 inputs are L2-resident)
        u4v alo = apl[(size_t)it * 4];
        u4v ahi = apl[(size_t)it * 4 + 1];
        BfCast ac;
        #pragma unroll
        for (int j = 0; j < 4; ++j) { ac.u[j] = alo[j]; ac.u[4 + j] = ahi[j]; }
        v16bf afrag = ac.b;

        const unsigned* bS32 = (const unsigned*)&bS[cur][0];
        const unsigned* cS32 = (const unsigned*)&cS[cur][0];
        #pragma unroll
        for (int nt = 0; nt < 2; ++nt) {
            int bb = (nt * 16 + n_low) * 16 + kb;
            u4v blo = *(const u4v*)(bS32 + bb);
            u4v bhi = *(const u4v*)(bS32 + bb + 4);
            u4v clo = *(const u4v*)(cS32 + bb);
            u4v chi = *(const u4v*)(cS32 + bb + 4);
            BfCast bc, cc;
            #pragma unroll
            for (int j = 0; j < 4; ++j) {
                bc.u[j] = blo[j]; bc.u[4 + j] = bhi[j];
                cc.u[j] = clo[j]; cc.u[4 + j] = chi[j];
            }
            accH[nt] = __builtin_amdgcn_wmma_f32_16x16x32_bf16(
                false, afrag, false, bc.b, (short)0, accH[nt], false, false);
            accA[nt] = __builtin_amdgcn_wmma_f32_16x16x32_bf16(
                false, afrag, false, cc.b, (short)0, accA[nt], false, false);
        }

        if (more) {                       // stage tile it+1 into the other buffer
            unsigned k0p = pack2bf(nw.x * nm.x, nw.y * nm.y);
            unsigned k1p = pack2bf(nw.z * nm.z, nw.w * nm.w);
            unsigned t0p = pack2bf(nt4.x, nt4.y);
            unsigned t1p = pack2bf(nt4.z, nt4.w);
            bS[nxt][(wn + 0) * KT + wk] = (unsigned short)k0p;
            bS[nxt][(wn + 1) * KT + wk] = (unsigned short)(k0p >> 16);
            bS[nxt][(wn + 2) * KT + wk] = (unsigned short)k1p;
            bS[nxt][(wn + 3) * KT + wk] = (unsigned short)(k1p >> 16);
            cS[nxt][(wn + 0) * KT + wk] = (unsigned short)t0p;
            cS[nxt][(wn + 1) * KT + wk] = (unsigned short)(t0p >> 16);
            cS[nxt][(wn + 2) * KT + wk] = (unsigned short)t1p;
            cS[nxt][(wn + 3) * KT + wk] = (unsigned short)(t1p >> 16);
        }
        // stores(it -> nxt) visible before iter it+1 reads them; also fences
        // reads(it, cur) against stores(it+1 -> cur).
        __syncthreads();
    }

    // ---- spill accumulators: C/D layout -> hS/tS (add bias into h) ----
    #pragma unroll
    for (int nt = 0; nt < 2; ++nt) {
        int col = nt * 16 + (lane & 15);
        float bv = bias[p0 + col];
        #pragma unroll
        for (int r = 0; r < 8; ++r) {
            int row = wave * 16 + (lane >> 4) * 8 + r;
            hS[row * NT + col] = accH[nt][r] + bv;
            tS[row * NT + col] = accA[nt][r];
        }
    }
    __syncthreads();

    // ---- batch statistics per pathway column (biased, as in reference) ----
    if (tid < NT) {
        int col = tid;
        float s = 0.f, ss = 0.f;
        for (int row = 0; row < BN; ++row) {
            float v = hS[row * NT + col];
            s += v; ss += v * v;
        }
        float mean = s * (1.0f / BN);
        float var  = ss * (1.0f / BN) - mean * mean;
        meanS[col] = mean;
        rstdS[col] = rsqrtf(var + BN_EPS);
    }
    __syncthreads();

    // ---- epilogue: BN + tanh + sigmoid gate + outputs + decision partials ----
    {
        int row = tid >> 1;
        int cb  = (tid & 1) * 16;
        float s = 0.f;
        #pragma unroll
        for (int j = 0; j < 16; ++j) {
            int col = cb + j;
            int p   = p0 + col;
            float h  = hS[row * NT + col];
            float hn = (h - meanS[col]) * rstdS[col] * gamma[p] + beta[p];
            float o  = fast_tanh(hn);
            float ap = fast_sigmoid(tS[row * NT + col] + attb[p]);
            float oc = o * ap;
            out[(size_t)row * PP + p]               = oc;   // outcome
            out[OUT_ATT_OFF + (size_t)row * PP + p] = ap;   // attention_probs
            s += oc * decw[p];
        }
        rowpart[tid] = s;
    }
    __syncthreads();
    if (tid < BN)
        wsdec[blockIdx.x * BN + tid] = rowpart[2 * tid] + rowpart[2 * tid + 1];
}

__global__ void __launch_bounds__(128)
bioxnet_decision(const float* __restrict__ wsdec,  // [P/NT, B]
                 const float* __restrict__ decb,   // [1]
                 float*       __restrict__ out)
{
    int b = threadIdx.x;
    float s = decb[0];
    for (int blk = 0; blk < PP / NT; ++blk)
        s += wsdec[blk * BN + b];
    out[OUT_DEC_OFF + b] = s;
}

extern "C" void kernel_launch(void* const* d_in, const int* in_sizes, int n_in,
                              void* d_out, int out_size, void* d_ws, size_t ws_size,
                              hipStream_t stream) {
    const float* inp   = (const float*)d_in[0];
    const float* mapp  = (const float*)d_in[1];
    const float* ker   = (const float*)d_in[2];
    const float* bias  = (const float*)d_in[3];
    const float* attk  = (const float*)d_in[4];
    const float* attb  = (const float*)d_in[5];
    const float* gamma = (const float*)d_in[6];
    const float* beta  = (const float*)d_in[7];
    const float* decw  = (const float*)d_in[8];
    const float* decb  = (const float*)d_in[9];
    float* out = (float*)d_out;

    float*          wsdec = (float*)d_ws;
    unsigned short* abf   = (unsigned short*)((char*)d_ws + WSDEC_ELEMS * sizeof(float));

    // inputs f32 -> bf16 once (5.1 MB, L2-resident for the GEMM)
    prep_inputs_bf16<<<dim3((BN * GG / 4 + 255) / 256), dim3(256), 0, stream>>>(
        inp, (unsigned*)abf);
    bioxnet_fused<<<dim3(PP / NT), dim3(256), 0, stream>>>(
        abf, mapp, ker, bias, attk, attb, gamma, beta, decw, out, wsdec);
    bioxnet_decision<<<dim3(1), dim3(128), 0, stream>>>(wsdec, decb, out);
}
